// MultiHead_68925635166687
// MI455X (gfx1250) — compile-verified, matrix operand order checked by reading, not run on hardware
//
#include <hip/hip_runtime.h>

// ---------------------------------------------------------------------------
// MultiHead edge attention for MI455X (gfx1250, wave32, WMMA bf16)
// E=800000 edges, C=64, H=4 heads, D=16, N=50000 nodes
// ---------------------------------------------------------------------------

typedef __attribute__((ext_vector_type(16))) __bf16 v16bf;
typedef __attribute__((ext_vector_type(8)))  float  v8f;
typedef __attribute__((ext_vector_type(4)))  float  f32x4;
typedef __attribute__((ext_vector_type(4)))  unsigned u32x4;

#define C_DIM 64
#define H_DIM 4
#define TILE_E 16
#define WAVES_PER_BLOCK 8
#define EDGES_PER_BLOCK (TILE_E * WAVES_PER_BLOCK)

union FragAB { v16bf v; __bf16 e[16]; };
union AccF   { v8f   v; float  f[8];  };

// --- LDS offset of a generic (flat) shared pointer: low 32 bits of flat addr
__device__ __forceinline__ unsigned lds_off_u32(const void* p) {
    return (unsigned)(unsigned long long)p;
}

// --- CDNA5 async global->LDS copy of 8 KB with 256 threads (2 x b128 each)
__device__ __forceinline__ void async_stage_8k(void* lds_dst, const void* gsrc) {
    unsigned t    = threadIdx.x;
    unsigned base = lds_off_u32(lds_dst);
#pragma unroll
    for (int r = 0; r < 2; ++r) {
        unsigned off = (t + r * 256u) * 16u;
        asm volatile("global_load_async_to_lds_b128 %0, %1, %2"
                     :: "v"(base + off), "v"(off), "s"(gsrc)
                     : "memory");
    }
}
__device__ __forceinline__ void async_wait0() {
    asm volatile("s_wait_asynccnt 0" ::: "memory");
}

// --- 16-lane (DPP row) all-reduce add: quad swaps + row rotations, pure VALU.
// After this every lane of each 16-lane row holds the row sum.
__device__ __forceinline__ float row_reduce_add(float v) {
    int x;
    x = __builtin_amdgcn_update_dpp(0, __float_as_int(v), 0xB1, 0xF, 0xF, true);  // quad_perm(1,0,3,2)
    v += __int_as_float(x);
    x = __builtin_amdgcn_update_dpp(0, __float_as_int(v), 0x4E, 0xF, 0xF, true);  // quad_perm(2,3,0,1)
    v += __int_as_float(x);
    x = __builtin_amdgcn_update_dpp(0, __float_as_int(v), 0x124, 0xF, 0xF, true); // row_ror:4
    v += __int_as_float(x);
    x = __builtin_amdgcn_update_dpp(0, __float_as_int(v), 0x128, 0xF, 0xF, true); // row_ror:8
    v += __int_as_float(x);
    return v;
}

// --- A fragment: 16x32 bf16 tile of a row-major f32 matrix row.
// Single-use streaming data -> non-temporal loads (TH=NT), keep L2 for gathers.
__device__ __forceinline__ v16bf load_a_frag(const float* __restrict__ row, int k0) {
    f32x4 f0 = __builtin_nontemporal_load((const f32x4*)(row + k0));
    f32x4 f1 = __builtin_nontemporal_load((const f32x4*)(row + k0) + 1);
    f32x4 f2 = __builtin_nontemporal_load((const f32x4*)(row + k0 + 16));
    f32x4 f3 = __builtin_nontemporal_load((const f32x4*)(row + k0 + 16) + 1);
    FragAB u;
#pragma unroll
    for (int i = 0; i < 4; ++i) {
        u.e[i]      = (__bf16)f0[i];
        u.e[4 + i]  = (__bf16)f1[i];
        u.e[8 + i]  = (__bf16)f2[i];
        u.e[12 + i] = (__bf16)f3[i];
    }
    return u.v;
}

// --- B fragment from LDS-resident W^T (bf16, [n][k] row-major 64x64).
__device__ __forceinline__ v16bf load_b_frag(const __bf16* __restrict__ sW, int n, int k0) {
    const __bf16* w = sW + n * 64 + k0;
    FragAB u;
#pragma unroll
    for (int i = 0; i < 8; ++i) { u.e[i] = w[i]; u.e[8 + i] = w[16 + i]; }
    return u.v;
}

// ---------------------------------------------------------------------------
// Kernel 0: init smax (0 bits == +0.0f, implements clamp(min=0) + empty segs)
//           + denom, and build bf16 transposed weights W^T[n][k] in workspace.
// ---------------------------------------------------------------------------
__global__ __launch_bounds__(256) void prep_kernel(
    const float* __restrict__ Wq, const float* __restrict__ Wk, const float* __restrict__ Wv,
    unsigned* __restrict__ smax_bits, float* __restrict__ denom,
    __bf16* __restrict__ wtq, __bf16* __restrict__ wtk, __bf16* __restrict__ wtv, int NH)
{
    int i = blockIdx.x * 256 + threadIdx.x;
    if (i < NH) { smax_bits[i] = 0u; denom[i] = 0.0f; }
    if (i < C_DIM * C_DIM) {
        int k = i >> 6, n = i & 63;
        wtq[n * 64 + k] = (__bf16)Wq[i];
        wtk[n * 64 + k] = (__bf16)Wk[i];
        wtv[n * 64 + k] = (__bf16)Wv[i];
    }
}

// ---------------------------------------------------------------------------
// Kernel 1: q/k projection via WMMA bf16, per-edge/head score via DPP row
//           reduction, run-compressed atomicMax into smax (uint-bit trick;
//           non-positive scores never beat the 0 floor, so skipped).
// ---------------------------------------------------------------------------
__global__ __launch_bounds__(256) void score_kernel(
    const float* __restrict__ Xq, const float* __restrict__ Xk,
    const __bf16* __restrict__ wtq, const __bf16* __restrict__ wtk,
    const float* __restrict__ bq, const float* __restrict__ bk,
    const int* __restrict__ index,
    float* __restrict__ score, unsigned* __restrict__ smax_bits, int E)
{
    __shared__ __align__(16) __bf16 sWq[64 * 64];
    __shared__ __align__(16) __bf16 sWk[64 * 64];
    async_stage_8k(sWq, wtq);
    async_stage_8k(sWk, wtk);
    async_wait0();
    __syncthreads();

    const int  lane = threadIdx.x & 31;
    const int  wave = threadIdx.x >> 5;
    const int  hf   = lane >> 4;        // wave half -> K-group / edge-row half
    const int  m    = lane & 15;        // A row within tile / B column within head
    const long e0   = (long)blockIdx.x * EDGES_PER_BLOCK + wave * TILE_E;
    if (e0 + TILE_E > E) return;        // wave-uniform (E % 128 == 0 anyway)

    const float* rowq = Xq + (e0 + m) * C_DIM;
    const float* rowk = Xk + (e0 + m) * C_DIM;
    __builtin_prefetch(rowq + EDGES_PER_BLOCK * C_DIM, 0, 1);
    __builtin_prefetch(rowk + EDGES_PER_BLOCK * C_DIM, 0, 1);

    v16bf aq[2], ak[2];
#pragma unroll
    for (int s = 0; s < 2; ++s) {
        int k0 = s * 32 + hf * 8;
        aq[s] = load_a_frag(rowq, k0);
        ak[s] = load_a_frag(rowk, k0);
    }

    // destination-node ids for this half's 8 edges (lane m==0 aggregates atomics)
    int idx8[8];
    if (m == 0) {
#pragma unroll
        for (int j = 0; j < 8; ++j) idx8[j] = index[e0 + j + hf * 8];
    }

#pragma unroll
    for (int h = 0; h < H_DIM; ++h) {
        AccF qa, ka;
#pragma unroll
        for (int j = 0; j < 8; ++j) { qa.f[j] = 0.0f; ka.f[j] = 0.0f; }
#pragma unroll
        for (int s = 0; s < 2; ++s) {
            int k0 = s * 32 + hf * 8;
            v16bf bqf = load_b_frag(sWq, h * 16 + m, k0);
            qa.v = __builtin_amdgcn_wmma_f32_16x16x32_bf16(
                false, aq[s], false, bqf, (short)0, qa.v, false, false);
            v16bf bkf = load_b_frag(sWk, h * 16 + m, k0);
            ka.v = __builtin_amdgcn_wmma_f32_16x16x32_bf16(
                false, ak[s], false, bkf, (short)0, ka.v, false, false);
        }
        float bqv = bq[h * 16 + m];
        float bkv = bk[h * 16 + m];

        // row-wise q.k dot: accumulator element (j,lane) = (M=j+8*hf, N=m);
        // reduce the products over N (16 lanes of each row) with DPP adds.
        float sj[8];
#pragma unroll
        for (int j = 0; j < 8; ++j)
            sj[j] = row_reduce_add((qa.f[j] + bqv) * (ka.f[j] + bkv)) * 0.25f; // 1/sqrt(16)

        // distributed score store: lane m (<8) writes edge e0 + m + 8*hf
        if (m < 8)
            score[(e0 + m + hf * 8) * H_DIM + h] = sj[m];

        // run-compressed atomicMax over sorted node ids (lane 0 / lane 16)
        if (m == 0) {
            int   curIdx = idx8[0];
            float curMax = sj[0];
#pragma unroll
            for (int j = 1; j < 8; ++j) {
                if (idx8[j] != curIdx) {
                    if (curMax > 0.0f)
                        atomicMax(&smax_bits[(long)curIdx * H_DIM + h],
                                  __float_as_uint(curMax));
                    curIdx = idx8[j];
                    curMax = sj[j];
                } else {
                    curMax = fmaxf(curMax, sj[j]);
                }
            }
            if (curMax > 0.0f)
                atomicMax(&smax_bits[(long)curIdx * H_DIM + h],
                          __float_as_uint(curMax));
        }
    }
}

// ---------------------------------------------------------------------------
// Kernel 2: denom[n,h] += exp(score - smax[index])
// Wave-level segmented reduction over the sorted keys before atomics:
// only run-leader lanes touch L2 (up to 8x fewer contended atomics).
// ---------------------------------------------------------------------------
__global__ __launch_bounds__(256) void denom_kernel(
    const float* __restrict__ score, const unsigned* __restrict__ smax_bits,
    const int* __restrict__ index, float* __restrict__ denom, int EH)
{
    int  i      = blockIdx.x * 256 + threadIdx.x;
    int  lane   = threadIdx.x & 31;
    bool active = i < EH;

    int   bi = -1;
    float v  = 0.0f;
    if (active) {
        int e = i >> 2;
        int h = i & 3;
        bi = index[e] * H_DIM + h;
        float sm = __uint_as_float(smax_bits[bi]);
        v = __expf(score[i] - sm);
    }

    // keys with equal value sit at stride 4 (same head, consecutive edges),
    // non-decreasing within the wave -> segmented shfl-down merge.
#pragma unroll
    for (int d = 4; d < 32; d <<= 1) {
        float ov = __shfl_down(v, d, 32);
        int   ob = __shfl_down(bi, d, 32);
        if ((lane + d) < 32 && ob == bi) v += ov;
    }
    int  bup    = __shfl_up(bi, 4, 32);
    bool leader = (lane < 4) || (bup != bi);
    if (active && leader)
        atomicAdd(&denom[bi], v);
}

// ---------------------------------------------------------------------------
// Kernel 3: v projection via WMMA, attn = exp(sc-sm)/(denom+exp(-sm)),
//           out = attn * v  (attn broadcast across the tile via width-16 shfl)
// ---------------------------------------------------------------------------
__global__ __launch_bounds__(256) void out_kernel(
    const float* __restrict__ Xv, const __bf16* __restrict__ wtv,
    const float* __restrict__ bv, const int* __restrict__ index,
    const float* __restrict__ score, const unsigned* __restrict__ smax_bits,
    const float* __restrict__ denom, float* __restrict__ out, int E)
{
    __shared__ __align__(16) __bf16 sWv[64 * 64];
    async_stage_8k(sWv, wtv);
    async_wait0();
    __syncthreads();

    const int  lane = threadIdx.x & 31;
    const int  wave = threadIdx.x >> 5;
    const int  hf   = lane >> 4;
    const int  m    = lane & 15;
    const long e0   = (long)blockIdx.x * EDGES_PER_BLOCK + wave * TILE_E;
    if (e0 + TILE_E > E) return;

    const float* rowv = Xv + (e0 + m) * C_DIM;
    __builtin_prefetch(rowv + EDGES_PER_BLOCK * C_DIM, 0, 1);

    v16bf av[2];
#pragma unroll
    for (int s = 0; s < 2; ++s) av[s] = load_a_frag(rowv, s * 32 + hf * 8);

    // each lane owns the softmax terms of edge e0+m: vectorized 16B gathers
    long  er   = e0 + m;
    int   idx  = index[er];
    f32x4 sc4  = *(const f32x4*)(score + er * H_DIM);
    u32x4 sm4  = *(const u32x4*)(smax_bits + (long)idx * H_DIM);
    f32x4 dn4  = *(const f32x4*)(denom + (long)idx * H_DIM);

#pragma unroll
    for (int h = 0; h < H_DIM; ++h) {
        AccF va;
#pragma unroll
        for (int j = 0; j < 8; ++j) va.f[j] = 0.0f;
#pragma unroll
        for (int s = 0; s < 2; ++s) {
            v16bf bvf = load_b_frag(sWv, h * 16 + m, s * 32 + hf * 8);
            va.v = __builtin_amdgcn_wmma_f32_16x16x32_bf16(
                false, av[s], false, bvf, (short)0, va.v, false, false);
        }
        float bvv = bv[h * 16 + m];
        float sm  = __uint_as_float(sm4[h]);
        float dn  = dn4[h] + __expf(-sm);
        float attn_all = __expf(sc4[h] - sm) / dn;   // attn of edge e0+m, head h

#pragma unroll
        for (int j = 0; j < 8; ++j) {
            // accumulator element (j,lane): row M = j + 8*hf, col n = m
            float aj = __shfl(attn_all, j + hf * 8, 16);
            long  e  = e0 + j + hf * 8;
            __builtin_nontemporal_store(aj * (va.f[j] + bvv),
                                        out + e * C_DIM + h * 16 + m);
        }
    }
}

// ---------------------------------------------------------------------------
extern "C" void kernel_launch(void* const* d_in, const int* in_sizes, int n_in,
                              void* d_out, int out_size, void* d_ws, size_t ws_size,
                              hipStream_t stream)
{
    const float* query = (const float*)d_in[0];
    const float* key_  = (const float*)d_in[1];
    const float* value = (const float*)d_in[2];
    const float* Wq    = (const float*)d_in[3];
    const float* bq    = (const float*)d_in[4];
    const float* Wk    = (const float*)d_in[5];
    const float* bk    = (const float*)d_in[6];
    const float* Wv    = (const float*)d_in[7];
    const float* bv    = (const float*)d_in[8];
    const int*   index = (const int*)d_in[9];

    const int E  = in_sizes[0] / C_DIM;   // 800000
    const int N  = 50000;                 // num_nodes (fixed by reference)
    const int EH = E * H_DIM;
    const int NH = N * H_DIM;

    // workspace: score[EH] | smax[NH] | denom[NH] | W^T bf16 x3  (16B aligned)
    char*     ws    = (char*)d_ws;
    float*    score = (float*)ws;
    unsigned* smaxb = (unsigned*)(ws + (size_t)EH * 4);
    float*    denom = (float*)(ws + (size_t)EH * 4 + (size_t)NH * 4);
    __bf16*   wtq   = (__bf16*)(ws + (size_t)EH * 4 + 2 * (size_t)NH * 4);
    __bf16*   wtk   = wtq + C_DIM * C_DIM;
    __bf16*   wtv   = wtk + C_DIM * C_DIM;

    prep_kernel<<<(NH + 255) / 256, 256, 0, stream>>>(
        Wq, Wk, Wv, smaxb, denom, wtq, wtk, wtv, NH);

    int blocks = (E + EDGES_PER_BLOCK - 1) / EDGES_PER_BLOCK;  // 6250
    score_kernel<<<blocks, 256, 0, stream>>>(
        query, key_, wtq, wtk, bq, bk, index, score, smaxb, E);

    denom_kernel<<<(EH + 255) / 256, 256, 0, stream>>>(
        score, smaxb, index, denom, EH);

    out_kernel<<<blocks, 256, 0, stream>>>(
        value, wtv, bv, index, score, smaxb, denom, (float*)d_out, E);
}
